// _RBFAnsatz_65481071400429
// MI455X (gfx1250) — matrix operand order repaired and checked by reading, not compiled, not run on hardware
//
#include <hip/hip_runtime.h>
#include <hip/hip_bf16.h>

// RBF kernel: out[n,m] = exp(-gamma * max(0, ||x_n||^2 + ||y_m||^2 - 2 x_n.y_m))
// N = M = 8192, D = 64, fp32 in/out. GEMM done with bf16x2 split-precision WMMA.

typedef __attribute__((ext_vector_type(16))) __bf16 v16bf;
typedef __attribute__((ext_vector_type(8)))  __bf16 v8bf;
typedef __attribute__((ext_vector_type(4)))  __bf16 v4bf;
typedef __attribute__((ext_vector_type(8)))  float  v8f;

#define NROWS 8192
#define MROWS 8192
#define DDIM  64
#define BT    128   // block tile (rows of x and rows of y per workgroup)
#define LDSP  72    // padded LDS row stride in bf16 elements (72*2 = 144 B)
#define GAMMA 1.0f

// Split one float4 into bf16 hi + bf16 residual-lo quads (RNE both times).
__device__ __forceinline__ void split4(const float4 v, v4bf& h, v4bf& l) {
  const float a0 = v.x, a1 = v.y, a2 = v.z, a3 = v.w;
  __bf16 h0 = (__bf16)a0, h1 = (__bf16)a1, h2 = (__bf16)a2, h3 = (__bf16)a3;
  h[0] = h0; h[1] = h1; h[2] = h2; h[3] = h3;
  l[0] = (__bf16)(a0 - (float)h0);
  l[1] = (__bf16)(a1 - (float)h1);
  l[2] = (__bf16)(a2 - (float)h2);
  l[3] = (__bf16)(a3 - (float)h3);
}

// A-matrix (16x32 bf16) fragment: lane half=0 holds K {kb+0..7, kb+16..23},
// half=1 holds K {kb+8..15, kb+24..31}; 2x 16B LDS loads.
__device__ __forceinline__ v16bf load_fragA(const __bf16* base, int row, int kb, int half) {
  union { v16bf v; v8bf h[2]; } f;
  const __bf16* p = base + row * LDSP + kb + half * 8;
  f.h[0] = *(const v8bf*)p;
  f.h[1] = *(const v8bf*)(p + 16);
  return f.v;
}

// B-matrix (32x16 bf16) fragment: lane half=0 holds K {kb+0..7, kb+16..23},
// half=1 holds K {kb+8..15, kb+24..31}; 2x 16B LDS loads.
__device__ __forceinline__ v16bf load_fragB(const __bf16* base, int col, int kb, int half) {
  union { v16bf v; v8bf h[2]; } f;
  const __bf16* p = base + col * LDSP + kb + half * 8;
  f.h[0] = *(const v8bf*)p;
  f.h[1] = *(const v8bf*)(p + 16);
  return f.v;
}

__global__ __launch_bounds__(256)
void rbf_bf16x2_wmma_kernel(const float* __restrict__ x,
                            const float* __restrict__ y,
                            float* __restrict__ out) {
  __shared__ __bf16 Xhi[BT][LDSP];
  __shared__ __bf16 Xlo[BT][LDSP];
  __shared__ __bf16 Yhi[BT][LDSP];
  __shared__ __bf16 Ylo[BT][LDSP];
  __shared__ float  sx2[BT];
  __shared__ float  sy2[BT];

  const int tid = threadIdx.x;
  const int bm  = blockIdx.x;   // column block -> y rows
  const int bn  = blockIdx.y;   // row block    -> x rows

  if (tid < BT) sx2[tid] = 0.0f; else sy2[tid - BT] = 0.0f;
  __syncthreads();

  // ---- Stage: fp32 -> (hi, lo) bf16 planes in LDS + exact fp32 row norms ----
  {
    const int srow = tid >> 1;          // 0..127 (two threads per row)
    const int scol = (tid & 1) << 5;    // 0 or 32
    const float* xg = x + (size_t)(bn * BT + srow) * DDIM + scol;
    const float* yg = y + (size_t)(bm * BT + srow) * DDIM + scol;
    float nx = 0.0f, ny = 0.0f;
#pragma unroll
    for (int i = 0; i < 8; ++i) {
      const float4 vx = ((const float4*)xg)[i];
      const float4 vy = ((const float4*)yg)[i];
      nx += vx.x * vx.x + vx.y * vx.y + vx.z * vx.z + vx.w * vx.w;
      ny += vy.x * vy.x + vy.y * vy.y + vy.z * vy.z + vy.w * vy.w;
      v4bf hx, lx, hy, ly;
      split4(vx, hx, lx);
      split4(vy, hy, ly);
      const int c = scol + i * 4;
      *(v4bf*)&Xhi[srow][c] = hx;
      *(v4bf*)&Xlo[srow][c] = lx;
      *(v4bf*)&Yhi[srow][c] = hy;
      *(v4bf*)&Ylo[srow][c] = ly;
    }
    atomicAdd(&sx2[srow], nx);   // ds_add_f32
    atomicAdd(&sy2[srow], ny);
  }
  __syncthreads();

  // ---- Compute: each wave owns 16 rows x 128 cols (8 tiles of 16x16) ----
  const int lane = tid & 31;
  const int wv   = tid >> 5;      // wave 0..7 -> row tile
  const int half = lane >> 4;     // 0 or 1
  const int r16  = lane & 15;

  v16bf ah[2], al[2];
#pragma unroll
  for (int k = 0; k < 2; ++k) {
    ah[k] = load_fragA(&Xhi[0][0], wv * 16 + r16, k * 32, half);
    al[k] = load_fragA(&Xlo[0][0], wv * 16 + r16, k * 32, half);
  }

  float rx2[8];
#pragma unroll
  for (int j = 0; j < 8; ++j) rx2[j] = sx2[wv * 16 + j + half * 8];

  const int nbase = bn * BT + wv * 16;

#pragma unroll
  for (int ct = 0; ct < 8; ++ct) {
    v8f acc = {};
#pragma unroll
    for (int k = 0; k < 2; ++k) {
      const v16bf bh = load_fragB(&Yhi[0][0], ct * 16 + r16, k * 32, half);
      const v16bf bl = load_fragB(&Ylo[0][0], ct * 16 + r16, k * 32, half);
      // dot += xhi.yhi + xhi.ylo + xlo.yhi  (bf16x2 split-precision)
      acc = __builtin_amdgcn_wmma_f32_16x16x32_bf16(false, ah[k], false, bh,
                                                    (short)0, acc, false, false);
      acc = __builtin_amdgcn_wmma_f32_16x16x32_bf16(false, ah[k], false, bl,
                                                    (short)0, acc, false, false);
      acc = __builtin_amdgcn_wmma_f32_16x16x32_bf16(false, al[k], false, bh,
                                                    (short)0, acc, false, false);
    }

    const int   m  = bm * BT + ct * 16 + r16;
    const float sy = sy2[ct * 16 + r16];
#pragma unroll
    for (int j = 0; j < 8; ++j) {
      float sq = rx2[j] + sy - 2.0f * acc[j];
      sq = fmaxf(sq, 0.0f);
      const float v = __expf(-GAMMA * sq);
      // Streamed 256 MB output: non-temporal store keeps L2 for the inputs.
      __builtin_nontemporal_store(
          v, out + (size_t)(nbase + j + half * 8) * (size_t)MROWS + m);
    }
  }
}

extern "C" void kernel_launch(void* const* d_in, const int* in_sizes, int n_in,
                              void* d_out, int out_size, void* d_ws, size_t ws_size,
                              hipStream_t stream) {
  (void)in_sizes; (void)n_in; (void)out_size; (void)d_ws; (void)ws_size;
  const float* x = (const float*)d_in[0];
  const float* y = (const float*)d_in[1];
  float* out = (float*)d_out;
  dim3 grid(MROWS / BT, NROWS / BT);   // 64 x 64 workgroups
  dim3 block(256);                     // 8 wave32 waves
  hipLaunchKernelGGL(rbf_bf16x2_wmma_kernel, grid, block, 0, stream, x, y, out);
}
